// SingleStandRwkvBlock_25305947308165
// MI455X (gfx1250) — matrix operand order repaired and checked by reading, not compiled, not run on hardware
//
#include <hip/hip_runtime.h>

typedef __bf16 bf16;
typedef __attribute__((ext_vector_type(16))) __bf16 v16bf;
typedef __attribute__((ext_vector_type(8)))  float  v8f;

constexpr int Hdim = 2048;
constexpr int Tdim = 4096;
constexpr int Idim = 8192;

// ------------------------------------------------------------------
// LayerNorm: one block per row, 256 threads
// ------------------------------------------------------------------
__global__ void ln_kernel(const float* __restrict__ in, const float* __restrict__ sc,
                          const float* __restrict__ bi, float* __restrict__ out) {
  __shared__ float s1[256], s2[256];
  const int tid = threadIdx.x;
  const size_t base = (size_t)blockIdx.x * Hdim;
  float s = 0.f, ss = 0.f;
  for (int h = tid; h < Hdim; h += 256) { float x = in[base + h]; s += x; ss += x * x; }
  s1[tid] = s; s2[tid] = ss; __syncthreads();
  for (int o = 128; o > 0; o >>= 1) {
    if (tid < o) { s1[tid] += s1[tid + o]; s2[tid] += s2[tid + o]; }
    __syncthreads();
  }
  const float mu   = s1[0] * (1.f / Hdim);
  const float var  = s2[0] * (1.f / Hdim) - mu * mu;
  const float rstd = rsqrtf(var + 1e-5f);
  for (int h = tid; h < Hdim; h += 256)
    out[base + h] = (in[base + h] - mu) * rstd * sc[h] + bi[h];
}

// ------------------------------------------------------------------
// Token-shift mix: bf16 outputs for the WMMA GEMMs
// ------------------------------------------------------------------
template<int NOUT>
__global__ void mix_kernel(const float* __restrict__ x,
                           const float* __restrict__ tk, const float* __restrict__ tv,
                           const float* __restrict__ tr,
                           bf16* __restrict__ ok, bf16* __restrict__ ov, bf16* __restrict__ orr) {
  const size_t idx = (size_t)blockIdx.x * 256 + threadIdx.x;
  const int h = (int)(idx & (Hdim - 1));
  const float xv = x[idx];
  const float c  = (idx >= (size_t)Hdim) ? x[idx - Hdim] : 0.f;  // token shift, zero init
  float a = tk[h]; ok[idx] = (bf16)(xv * a + c * (1.f - a));
  if (NOUT == 3) { a = tv[h]; ov[idx] = (bf16)(xv * a + c * (1.f - a)); }
  a = tr[h]; orr[idx] = (bf16)(xv * a + c * (1.f - a));
}

// ------------------------------------------------------------------
// WKV recurrence: one thread per channel, serial over T.
// Fuses r*wkv and the bf16 conversion for the Wo GEMM.
// ------------------------------------------------------------------
__global__ void wkv_kernel(const float* __restrict__ kf, const float* __restrict__ vf,
                           const float* __restrict__ rf,
                           const float* __restrict__ time_decay,
                           const float* __restrict__ time_first,
                           bf16* __restrict__ out) {
  const int h = blockIdx.x * blockDim.x + threadIdx.x;
  const float decay = -expf(time_decay[h]);
  const float tf    = time_first[h];
  float aa = 0.f, bb = 0.f, pp = -__builtin_inff();
  size_t idx = (size_t)h;
  for (int t = 0; t < Tdim; ++t, idx += Hdim) {
    const float kk = kf[idx], vv = vf[idx];
    const float ww = tf + kk;
    const float p  = fmaxf(pp, ww);
    const float e1 = expf(pp - p), e2 = expf(ww - p);
    const float ot = (e1 * aa + e2 * vv) / (e1 * bb + e2);
    const float ww2 = decay + pp;
    const float p2  = fmaxf(ww2, kk);
    const float e1b = expf(ww2 - p2), e2b = expf(kk - p2);
    aa = e1b * aa + e2b * vv;
    bb = e1b * bb + e2b;
    pp = p2;
    out[idx] = (bf16)(rf[idx] * ot);
  }
}

// ------------------------------------------------------------------
// Weight convert fp32 -> bf16 with transpose: Wt[n*K + k] = W[k*N + n]
// ------------------------------------------------------------------
__global__ void cvt_t_kernel(const float* __restrict__ W, bf16* __restrict__ Wt,
                             int Kd, int Nd) {
  const size_t total = (size_t)Kd * Nd;
  for (size_t i = (size_t)blockIdx.x * 256 + threadIdx.x; i < total;
       i += (size_t)gridDim.x * 256) {
    const int k = (int)(i / (size_t)Nd);
    const int n = (int)(i - (size_t)k * Nd);
    Wt[(size_t)n * Kd + k] = (bf16)W[i];
  }
}

// ------------------------------------------------------------------
// WMMA GEMM: C[M,N] = A[M,K](bf16 row-major) x B (bf16 pre-transposed
// to Bt[N,K]), fp32 accum via v_wmma_f32_16x16x32_bf16.
// Block tile 128x128, BK=64, double-buffered LDS filled with
// GLOBAL_LOAD_ASYNC_TO_LDS_B128 (ASYNCcnt). 8 waves, each 32x64.
// ------------------------------------------------------------------
enum { EPI_NONE = 0, EPI_SIGMOID = 1, EPI_RESADD = 2, EPI_RELUSQ_BF16 = 3, EPI_FINAL = 4 };

__device__ __forceinline__ void async_b128(unsigned lds_addr, int goff, uint64_t gbase) {
  // GVS form: LDS-dest VGPR, 32-bit byte-offset VGPR, 64-bit SGPR base
  asm volatile("global_load_async_to_lds_b128 %0, %1, %2"
               :: "v"(lds_addr), "v"(goff), "s"(gbase) : "memory");
}
__device__ __forceinline__ void wait_async0() {
  asm volatile("s_wait_asynccnt 0" ::: "memory");
}

template<int EPI>
__launch_bounds__(256, 2)
__global__ void gemm_wmma_kernel(const bf16* __restrict__ A, const bf16* __restrict__ Bt,
                                 void* __restrict__ Cout,
                                 const float* __restrict__ res, const float* __restrict__ scl,
                                 int M, int N, int K) {
  constexpr int BM = 128, BN = 128, BK = 64;
  constexpr int LDT = BK + 8;                 // padded stride (144B, 16B-aligned)
  __shared__ bf16 As[2][BM * LDT];
  __shared__ bf16 Bs[2][BN * LDT];

  const int tid  = threadIdx.x;
  const int lane = tid & 31;
  const int wid  = tid >> 5;
  const int wm0  = (wid >> 1) * 32;           // 4 waves along M
  const int wn0  = (wid & 1) * 64;            // 2 waves along N
  const int m0   = blockIdx.y * BM;
  const int n0   = blockIdx.x * BN;

  const int srow = tid >> 3;                  // 0..31
  const int scol = (tid & 7) << 3;            // 0..56 step 8 (16B chunks)
  const uint64_t abase = (uint64_t)A;
  const uint64_t bbase = (uint64_t)Bt;

  auto stage = [&](int buf, int kb) {
#pragma unroll
    for (int i = 0; i < 4; ++i) {
      const int row = srow + i * 32;
      const unsigned la = (unsigned)(uintptr_t)&As[buf][row * LDT + scol];
      const int ga = (int)(((size_t)(m0 + row) * K + kb + scol) * 2);
      async_b128(la, ga, abase);
      const unsigned lb = (unsigned)(uintptr_t)&Bs[buf][row * LDT + scol];
      const int gb = (int)(((size_t)(n0 + row) * K + kb + scol) * 2);
      async_b128(lb, gb, bbase);
    }
  };

  v8f acc[2][4] = {};

  stage(0, 0);
  wait_async0();
  __syncthreads();

  int cur = 0;
  for (int kb = 0; kb < K; kb += BK) {
    if (kb + BK < K) stage(cur ^ 1, kb + BK);     // async prefetch of next tile
    // L2 warm-up hint two tiles ahead (global_prefetch_b8)
    const int pk = (kb + 2 * BK < K) ? (kb + 2 * BK) : kb;
    __builtin_prefetch(A  + (size_t)(m0 + srow) * K + pk + scol, 0, 1);
    __builtin_prefetch(Bt + (size_t)(n0 + srow) * K + pk + scol, 0, 1);

    const int ml   = lane & 15;
    const int half = lane >> 4;
#pragma unroll
    for (int ks = 0; ks < BK; ks += 32) {
      union { v16bf v; uint4 q[2]; } a[2], b[4];
      // A fragment (16x32): lane m=ml; K {half*8..+7} then {16+half*8..+7}
#pragma unroll
      for (int mi = 0; mi < 2; ++mi) {
        const bf16* p = &As[cur][(wm0 + mi * 16 + ml) * LDT + ks + half * 8];
        a[mi].q[0] = *(const uint4*)p;
        a[mi].q[1] = *(const uint4*)(p + 16);
      }
      // B fragment (32x16): lane n=ml; K {half*8..+7} then {16+half*8..+7}
#pragma unroll
      for (int ni = 0; ni < 4; ++ni) {
        const bf16* p = &Bs[cur][(wn0 + ni * 16 + ml) * LDT + ks + half * 8];
        b[ni].q[0] = *(const uint4*)p;
        b[ni].q[1] = *(const uint4*)(p + 16);
      }
#pragma unroll
      for (int mi = 0; mi < 2; ++mi)
#pragma unroll
        for (int ni = 0; ni < 4; ++ni)
          acc[mi][ni] = __builtin_amdgcn_wmma_f32_16x16x32_bf16(
              false, a[mi].v, false, b[ni].v, (short)0, acc[mi][ni], false, false);
    }
    wait_async0();          // next-tile async copies landed
    __syncthreads();        // everyone done reading cur / writing cur^1
    cur ^= 1;
  }

  // Epilogue. C layout: VGPR r -> M = r + 8*(lane>>4); N = lane&15.
  const int rbase = (lane >> 4) * 8;
  const int cn    = lane & 15;
#pragma unroll
  for (int mi = 0; mi < 2; ++mi) {
#pragma unroll
    for (int ni = 0; ni < 4; ++ni) {
#pragma unroll
      for (int r = 0; r < 8; ++r) {
        const int row = m0 + wm0 + mi * 16 + rbase + r;
        const int col = n0 + wn0 + ni * 16 + cn;
        const size_t idx = (size_t)row * N + col;
        float v = acc[mi][ni][r];
        if (EPI == EPI_NONE) {
          ((float*)Cout)[idx] = v;
        } else if (EPI == EPI_SIGMOID) {
          ((float*)Cout)[idx] = 1.f / (1.f + expf(-v));
        } else if (EPI == EPI_RESADD) {
          ((float*)Cout)[idx] = res[idx] + v;
        } else if (EPI == EPI_RELUSQ_BF16) {
          v = fmaxf(v, 0.f);
          ((bf16*)Cout)[idx] = (bf16)(v * v);
        } else {  // EPI_FINAL: hidden2 + rr * acc
          ((float*)Cout)[idx] = res[idx] + scl[idx] * v;
        }
      }
    }
  }
}

// ------------------------------------------------------------------
extern "C" void kernel_launch(void* const* d_in, const int* in_sizes, int n_in,
                              void* d_out, int out_size, void* d_ws, size_t ws_size,
                              hipStream_t stream) {
  const float* hidden     = (const float*)d_in[0];
  const float* ln1_s      = (const float*)d_in[1];
  const float* ln1_b      = (const float*)d_in[2];
  const float* ln2_s      = (const float*)d_in[3];
  const float* ln2_b      = (const float*)d_in[4];
  const float* time_decay = (const float*)d_in[5];
  const float* time_first = (const float*)d_in[6];
  const float* tm_k       = (const float*)d_in[7];
  const float* tm_v       = (const float*)d_in[8];
  const float* tm_r       = (const float*)d_in[9];
  const float* Wk         = (const float*)d_in[10];
  const float* Wv         = (const float*)d_in[11];
  const float* Wr         = (const float*)d_in[12];
  const float* Wo         = (const float*)d_in[13];
  const float* ffn_tm_k   = (const float*)d_in[14];
  const float* ffn_tm_r   = (const float*)d_in[15];
  const float* Wk_ff      = (const float*)d_in[16];
  const float* Wv_ff      = (const float*)d_in[17];
  const float* Wr_ff      = (const float*)d_in[18];

  const size_t TH_f = (size_t)Tdim * Hdim * 4;
  const size_t TH_b = (size_t)Tdim * Hdim * 2;
  const size_t HH_b = (size_t)Hdim * Hdim * 2;
  const size_t HI_b = (size_t)Hdim * Idim * 2;
  const size_t TI_b = (size_t)Tdim * Idim * 2;

  char* wsp = (char*)d_ws;
  size_t off = 0;
  auto alloc = [&](size_t bytes) -> void* {
    void* p = wsp + off;
    off += (bytes + 255) & ~(size_t)255;
    return p;
  };
  float* x_ln    = (float*)alloc(TH_f);
  bf16*  kxb     = (bf16*) alloc(TH_b);   // key_x (later xk)
  bf16*  vxb     = (bf16*) alloc(TH_b);   // value_x
  bf16*  rxb     = (bf16*) alloc(TH_b);   // recep_x (later xr)
  float* kf      = (float*)alloc(TH_f);
  float* vf      = (float*)alloc(TH_f);
  float* rf      = (float*)alloc(TH_f);   // sigmoid(r) (later rr)
  bf16*  rwkvb   = (bf16*) alloc(TH_b);   // r * wkv, bf16
  float* hidden2 = (float*)alloc(TH_f);
  bf16*  Wk_t    = (bf16*) alloc(HH_b);
  bf16*  Wv_t    = (bf16*) alloc(HH_b);
  bf16*  Wr_t    = (bf16*) alloc(HH_b);
  bf16*  Wo_t    = (bf16*) alloc(HH_b);
  bf16*  Wrff_t  = (bf16*) alloc(HH_b);
  bf16*  Wkff_t  = (bf16*) alloc(HI_b);   // [I][H]
  bf16*  Wvff_t  = (bf16*) alloc(HI_b);   // [H][I]
  bf16*  kkb     = (bf16*) alloc(TI_b);   // relu(xk@Wk_ff)^2, bf16

  // weights: fp32 -> bf16, transposed to [N][K] for K-major LDS staging
  cvt_t_kernel<<<4096, 256, 0, stream>>>(Wk,    Wk_t,   Hdim, Hdim);
  cvt_t_kernel<<<4096, 256, 0, stream>>>(Wv,    Wv_t,   Hdim, Hdim);
  cvt_t_kernel<<<4096, 256, 0, stream>>>(Wr,    Wr_t,   Hdim, Hdim);
  cvt_t_kernel<<<4096, 256, 0, stream>>>(Wo,    Wo_t,   Hdim, Hdim);
  cvt_t_kernel<<<4096, 256, 0, stream>>>(Wr_ff, Wrff_t, Hdim, Hdim);
  cvt_t_kernel<<<8192, 256, 0, stream>>>(Wk_ff, Wkff_t, Hdim, Idim);
  cvt_t_kernel<<<8192, 256, 0, stream>>>(Wv_ff, Wvff_t, Idim, Hdim);

  // ---- time-mix ----
  ln_kernel<<<Tdim, 256, 0, stream>>>(hidden, ln1_s, ln1_b, x_ln);
  mix_kernel<3><<<(Tdim * Hdim) / 256, 256, 0, stream>>>(x_ln, tm_k, tm_v, tm_r,
                                                         kxb, vxb, rxb);
  dim3 blk(256);
  dim3 gHH(Hdim / 128, Tdim / 128);
  gemm_wmma_kernel<EPI_NONE>   <<<gHH, blk, 0, stream>>>(kxb, Wk_t, kf, nullptr, nullptr, Tdim, Hdim, Hdim);
  gemm_wmma_kernel<EPI_NONE>   <<<gHH, blk, 0, stream>>>(vxb, Wv_t, vf, nullptr, nullptr, Tdim, Hdim, Hdim);
  gemm_wmma_kernel<EPI_SIGMOID><<<gHH, blk, 0, stream>>>(rxb, Wr_t, rf, nullptr, nullptr, Tdim, Hdim, Hdim);

  wkv_kernel<<<Hdim / 256, 256, 0, stream>>>(kf, vf, rf, time_decay, time_first, rwkvb);

  gemm_wmma_kernel<EPI_RESADD><<<gHH, blk, 0, stream>>>(rwkvb, Wo_t, hidden2, hidden, nullptr, Tdim, Hdim, Hdim);

  // ---- channel-mix ----
  ln_kernel<<<Tdim, 256, 0, stream>>>(hidden2, ln2_s, ln2_b, x_ln);
  mix_kernel<2><<<(Tdim * Hdim) / 256, 256, 0, stream>>>(x_ln, ffn_tm_k, nullptr, ffn_tm_r,
                                                         kxb, nullptr, rxb);

  gemm_wmma_kernel<EPI_SIGMOID><<<gHH, blk, 0, stream>>>(rxb, Wrff_t, rf, nullptr, nullptr, Tdim, Hdim, Hdim);
  dim3 gHI(Idim / 128, Tdim / 128);
  gemm_wmma_kernel<EPI_RELUSQ_BF16><<<gHI, blk, 0, stream>>>(kxb, Wkff_t, kkb, nullptr, nullptr, Tdim, Idim, Hdim);
  gemm_wmma_kernel<EPI_FINAL><<<gHH, blk, 0, stream>>>(kkb, Wvff_t, (float*)d_out, hidden2, rf, Tdim, Hdim, Idim);
}
